// DenseKANet_62148176773778
// MI455X (gfx1250) — matrix operand order repaired and checked by reading, not compiled, not run on hardware
//
#include <hip/hip_runtime.h>
#include <hip/hip_bf16.h>

// ---------------------------------------------------------------------------
// DenseKANet on MI455X (gfx1250): every FastKAN conv lowered to ONE implicit
// GEMM (K = (Cin + 8*Cin)*k*k, base+spline concatenated along K) executed by
// v_wmma_f32_16x16x32_f16.  GEMM: 64x128x32 block tile, 8 waves each owning a
// 32x32 C macro-tile (4 WMMAs / K-step), triple-buffered LDS staging fed by
// GLOBAL_LOAD_ASYNC_TO_LDS_B128 (ASYNCcnt-tracked) when the toolchain
// declares the builtin, else sync global_load+ds_store.  Weights are packed
// f16 zero-padded to M%64==0 / K%32==0 so all tile loads are unguarded.
// ---------------------------------------------------------------------------

typedef _Float16 h8   __attribute__((ext_vector_type(8)));
typedef _Float16 v16h __attribute__((ext_vector_type(16)));
typedef float    v8f  __attribute__((ext_vector_type(8)));

#ifndef __has_builtin
#define __has_builtin(x) 0
#endif
#if __has_builtin(__builtin_amdgcn_global_load_async_to_lds_b128) && \
    __has_builtin(__builtin_amdgcn_s_wait_asynccnt)
#define HAVE_ASYNC 1
#else
#define HAVE_ASYNC 0
#endif

#if HAVE_ASYNC
// Builtin signature (from hipcc diagnostic): both pointers are to a 16-byte
// int vector; src in AS(1) ("__device__"), dst in AS(3) ("__shared__").
typedef int i4 __attribute__((vector_size(16)));
typedef __attribute__((address_space(1))) i4 as1_i4;
typedef __attribute__((address_space(3))) i4 as3_i4;
__device__ __forceinline__ as1_i4* gcast(const void* p) {
  return (as1_i4*)(unsigned long long)(__SIZE_TYPE__)p;
}
__device__ __forceinline__ as3_i4* lcast(const void* p) {
  // generic LDS pointer: low 32 bits are the LDS byte offset (ISA aperture rule)
  return (as3_i4*)(unsigned int)(__SIZE_TYPE__)p;
}
#endif

#define PIXCHUNK 1024   // multiple of BN
#define BM 64
#define BN 128
#define BK 32
#define KLD 40          // LDS row stride in halves (80B, 16B-aligned)
#define NBUF 3

// ------------------------------- elementwise -------------------------------

__global__ void k_silu(const float* __restrict__ in, int inBS,
                       float* __restrict__ out, int CHW, int n) {
  int idx = blockIdx.x * blockDim.x + threadIdx.x;
  if (idx >= n) return;
  int b = idx / CHW;
  int rem = idx - b * CHW;
  float v = in[(size_t)b * inBS + rem];
  out[idx] = v / (1.f + __expf(-v));
}

__global__ void k_instnorm_stats(const float* __restrict__ in, int inBS,
                                 float* __restrict__ mean, float* __restrict__ rstd,
                                 int C, int HW) {
  int bc = blockIdx.x;
  int b = bc / C, c = bc - b * C;
  const float* p = in + (size_t)b * inBS + (size_t)c * HW;
  float s = 0.f, s2 = 0.f;
  for (int i = threadIdx.x; i < HW; i += blockDim.x) {
    float v = p[i]; s += v; s2 += v * v;
  }
  __shared__ float sh1[256], sh2[256];
  sh1[threadIdx.x] = s; sh2[threadIdx.x] = s2;
  __syncthreads();
  for (int off = 128; off > 0; off >>= 1) {
    if ((int)threadIdx.x < off) {
      sh1[threadIdx.x] += sh1[threadIdx.x + off];
      sh2[threadIdx.x] += sh2[threadIdx.x + off];
    }
    __syncthreads();
  }
  if (threadIdx.x == 0) {
    float m = sh1[0] / (float)HW;
    float v = sh2[0] / (float)HW - m * m;
    mean[bc] = m;
    rstd[bc] = rsqrtf(v + 1e-5f);
  }
}

__global__ void k_rbf(const float* __restrict__ in, int inBS,
                      const float* __restrict__ mean, const float* __restrict__ rstd,
                      float* __restrict__ out, int C, int HW, int n) {
  int idx = blockIdx.x * blockDim.x + threadIdx.x;
  if (idx >= n) return;
  int hw = idx % HW;
  int t = idx / HW;
  int c = t % C;
  int b = t / C;
  float xn = (in[(size_t)b * inBS + (size_t)c * HW + hw] - mean[b * C + c]) * rstd[b * C + c];
  const float denom = 4.0f / 7.0f;   // 4/(GRID_SIZE-1); grid spacing equals denom
  float* o = out + ((size_t)(b * C + c) * 8) * HW + hw;
#pragma unroll
  for (int g = 0; g < 8; ++g) {
    float gv = -2.0f + (float)g * denom;
    float tt = (xn - gv) / denom;
    o[(size_t)g * HW] = __expf(-tt * tt);
  }
}

__global__ void k_bn_selu(const float* __restrict__ in,
                          const float* __restrict__ gam, const float* __restrict__ bet,
                          const float* __restrict__ mn,  const float* __restrict__ vr,
                          float* __restrict__ out, int C, int HW, int n) {
  int idx = blockIdx.x * blockDim.x + threadIdx.x;
  if (idx >= n) return;
  int c = (idx / HW) % C;
  float scale = gam[c] * rsqrtf(vr[c] + 1e-5f);
  float y = (in[idx] - mn[c]) * scale + bet[c];
  const float a = 1.6732632423543772f, sc = 1.0507009873554805f;
  out[idx] = sc * (y > 0.f ? y : a * (__expf(y) - 1.f));
}

__global__ void k_maxpool3(const float* __restrict__ in, float* __restrict__ out,
                           int outBS, int C, int H, int W, int n) {
  int idx = blockIdx.x * blockDim.x + threadIdx.x;
  if (idx >= n) return;
  int OH = (H - 1) / 2 + 1, OW = (W - 1) / 2 + 1;  // k=3,s=2,p=1
  int ow = idx % OW; int t = idx / OW;
  int oh = t % OH; t /= OH;
  int c = t % C; int b = t / C;
  float m = -3.4e38f;
#pragma unroll
  for (int kh = 0; kh < 3; ++kh) {
    int ih = oh * 2 - 1 + kh;
    if (ih < 0 || ih >= H) continue;
#pragma unroll
    for (int kw = 0; kw < 3; ++kw) {
      int iw = ow * 2 - 1 + kw;
      if (iw < 0 || iw >= W) continue;
      m = fmaxf(m, in[(((size_t)b * C + c) * H + ih) * W + iw]);
    }
  }
  out[(size_t)b * outBS + ((size_t)c * OH + oh) * OW + ow] = m;
}

__global__ void k_avgpool2(const float* __restrict__ in, float* __restrict__ out,
                           int outBS, int C, int H, int W, int n) {
  int idx = blockIdx.x * blockDim.x + threadIdx.x;
  if (idx >= n) return;
  int OH = H / 2, OW = W / 2;
  int ow = idx % OW; int t = idx / OW;
  int oh = t % OH; t /= OH;
  int c = t % C; int b = t / C;
  const float* p = in + (((size_t)b * C + c) * H + oh * 2) * W + ow * 2;
  float s = p[0] + p[1] + p[W] + p[W + 1];
  out[(size_t)b * outBS + ((size_t)c * OH + oh) * OW + ow] = 0.25f * s;
}

__global__ void k_gap(const float* __restrict__ in, float* __restrict__ out, int HW) {
  int bc = blockIdx.x;
  const float* p = in + (size_t)bc * HW;
  float s = 0.f;
  for (int i = threadIdx.x; i < HW; i += blockDim.x) s += p[i];
  __shared__ float sh[64];
  sh[threadIdx.x] = s;
  __syncthreads();
  for (int off = 32; off > 0; off >>= 1) {
    if ((int)threadIdx.x < off) sh[threadIdx.x] += sh[threadIdx.x + off];
    __syncthreads();
  }
  if (threadIdx.x == 0) out[bc] = sh[0] / (float)HW;
}

__global__ void k_fc(const float* __restrict__ feat, const float* __restrict__ Wm,
                     const float* __restrict__ bias, float* __restrict__ out,
                     int Bn, int M, int K) {
  int idx = blockIdx.x * blockDim.x + threadIdx.x;
  if (idx >= Bn * M) return;
  int b = idx / M, m = idx - b * M;
  const float* f = feat + (size_t)b * K;
  const float* w = Wm + (size_t)m * K;
  float s = bias[m];
  for (int k = 0; k < K; ++k) s += f[k] * w[k];
  out[(size_t)b * M + m] = s;
}

// --------------------------- GEMM staging kernels ---------------------------

// Pack [base_w | spline_w] -> f16 row-major [Mpad][Kpad], zero-padded in both
// M (to %64) and K (to %32) so GEMM tile loads need no guards.
__global__ void k_pack_w(const float* __restrict__ bw, const float* __restrict__ sw,
                         _Float16* __restrict__ w16, int cout, int Cb, int Cs,
                         int ksz2, int K, int Kpad) {
  int kk = blockIdx.x * blockDim.x + threadIdx.x;
  if (kk >= Kpad) return;
  int co = blockIdx.y;                  // 0..Mpad-1
  float v = 0.f;
  if (co < cout && kk < K) {
    int baseK = Cb * ksz2;
    if (kk < baseK) v = bw[(size_t)co * baseK + kk];
    else            v = sw[(size_t)co * ((size_t)Cs * ksz2) + (kk - baseK)];
  }
  w16[(size_t)co * Kpad + kk] = (_Float16)v;
}

// im2col (base + spline concatenated along K) -> f16 [chunkN][Kpad].
__global__ void k_im2col(const float* __restrict__ actB, const float* __restrict__ actS,
                         _Float16* __restrict__ col, int p0, int Cb, int Cs,
                         int H, int W, int OW, int OHW,
                         int kd, int stride, int pad, int K, int Kpad) {
  int kk = blockIdx.x * blockDim.x + threadIdx.x;
  if (kk >= Kpad) return;
  int pl = blockIdx.y;
  float v = 0.f;
  if (kk < K) {
    int ksz2 = kd * kd;
    int baseK = Cb * ksz2;
    const float* act;
    int c, r, Cact;
    if (kk < baseK) { act = actB; Cact = Cb; c = kk / ksz2; r = kk - c * ksz2; }
    else { int k2 = kk - baseK; act = actS; Cact = Cs; c = k2 / ksz2; r = k2 - c * ksz2; }
    int kh = r / kd, kw = r - kh * kd;
    int p = p0 + pl;
    int b = p / OHW;
    int rem = p - b * OHW;
    int oh = rem / OW, ow = rem - (rem / OW) * OW;
    int ih = oh * stride - pad + kh;
    int iw = ow * stride - pad + kw;
    if (ih >= 0 && ih < H && iw >= 0 && iw < W)
      v = act[(((size_t)b * Cact + c) * H + ih) * W + iw];
  }
  col[(size_t)pl * Kpad + kk] = (_Float16)v;
}

// ------------------------------- WMMA GEMM ----------------------------------
// C[Mpad][N] = A[Mpad][Kpad] * B^T (B stored [N][Kpad]).  8 waves / block in a
// 2x4 grid; each wave owns a 32x32 C macro-tile -> 4 v_wmma per K-step from
// 2 A-frags x 2 B-frags.  Triple-buffered LDS, one barrier per K-step, tiles
// fed by GLOBAL_LOAD_ASYNC_TO_LDS_B128 (s_wait_asynccnt) when available.
// Epilogue scatters NCHW: out[b*outBS + co*OHW + rem], p = p0 + n.
__global__ __launch_bounds__(256)
void k_gemm_wmma(const _Float16* __restrict__ A, const _Float16* __restrict__ Bmat,
                 float* __restrict__ outBase, int M, int N, int Kpad,
                 int p0, int OHW, int outBS, int Ntot) {
  __shared__ _Float16 As[NBUF][BM][KLD];
  __shared__ _Float16 Bs[NBUF][BN][KLD];

  const int tid  = threadIdx.x;
  const int lane = tid & 31;
  const int wv   = tid >> 5;
  const int mt   = wv >> 2;      // 0..1  (32-row macro-tile)
  const int ntw  = wv & 3;       // 0..3  (32-col macro-tile)
  const int m0   = blockIdx.y * BM;
  const int n0   = blockIdx.x * BN;

  const v8f vzero = {0.f, 0.f, 0.f, 0.f, 0.f, 0.f, 0.f, 0.f};
  v8f acc00 = vzero, acc01 = vzero, acc10 = vzero, acc11 = vzero;

  // One K-slice: A 64x32 halves (256 thr x 16B), B 128x32 halves (256 thr x 2x16B)
  auto issueTile = [&](int kb, int buf) {
    int arow = tid >> 2;
    int akc  = (tid & 3) * 8;
#if HAVE_ASYNC
    __builtin_amdgcn_global_load_async_to_lds_b128(
        gcast(A + (size_t)(m0 + arow) * Kpad + kb + akc),
        lcast(&As[buf][arow][akc]), 0, 0);
#pragma unroll
    for (int q = 0; q < 2; ++q) {
      int cid = tid + q * 256;
      int brow = cid >> 2;
      int bkc  = (cid & 3) * 8;
      __builtin_amdgcn_global_load_async_to_lds_b128(
          gcast(Bmat + (size_t)(n0 + brow) * Kpad + kb + bkc),
          lcast(&Bs[buf][brow][bkc]), 0, 0);
    }
#else
    const _Float16* asrc = A + (size_t)(m0 + arow) * Kpad + kb + akc;
    h8 av = *(const h8*)asrc;
    __builtin_prefetch(asrc + BK, 0, 0);   // -> global_prefetch_b8
    *(h8*)&As[buf][arow][akc] = av;
#pragma unroll
    for (int q = 0; q < 2; ++q) {
      int cid = tid + q * 256;
      int brow = cid >> 2;
      int bkc  = (cid & 3) * 8;
      const _Float16* bsrc = Bmat + (size_t)(n0 + brow) * Kpad + kb + bkc;
      h8 bv = *(const h8*)bsrc;
      __builtin_prefetch(bsrc + BK, 0, 0);
      *(h8*)&Bs[buf][brow][bkc] = bv;
    }
#endif
  };

  // A fragment (16x32 f16, ISA 7.12.2): lanes 0-15 hold K0-7 & K16-23,
  // lanes 16-31 hold K8-15 & K24-31.
  auto ldA = [&](int buf, int r, int k8) -> v16h {
    h8 lo = *(const h8*)&As[buf][r][k8];
    h8 hi = *(const h8*)&As[buf][r][k8 + 16];
    v16h f;
#pragma unroll
    for (int i = 0; i < 8; ++i) { f[i] = lo[i]; f[i + 8] = hi[i]; }
    return f;
  };
  // B fragment (32x16 f16): lanes 0-15 hold K0-7 & K16-23, lanes 16-31 hold K8-15 & K24-31; lane%16 = column.
  auto ldB = [&](int buf, int r, int k16) -> v16h {
    h8 lo = *(const h8*)&Bs[buf][r][k16];
    h8 hi = *(const h8*)&Bs[buf][r][k16 + 16];
    v16h f;
#pragma unroll
    for (int i = 0; i < 8; ++i) { f[i] = lo[i]; f[i + 8] = hi[i]; }
    return f;
  };

  issueTile(0, 0);

  const int nk = Kpad / BK;
  for (int kt = 0; kt < nk; ++kt) {
    int cur = kt % NBUF;
    if (kt + 1 < nk) {
      issueTile((kt + 1) * BK, (kt + 1) % NBUF);
#if HAVE_ASYNC
      __builtin_amdgcn_s_wait_asynccnt(3);   // slice-kt's 3 ops complete
#endif
    } else {
#if HAVE_ASYNC
      __builtin_amdgcn_s_wait_asynccnt(0);
#endif
    }
    __syncthreads();

    int ar = mt * 32 + (lane & 15);
    int ak = (lane >> 4) * 8;
    v16h a0 = ldA(cur, ar, ak);
    v16h a1 = ldA(cur, ar + 16, ak);
    int bc = ntw * 32 + (lane & 15);
    int bk = (lane >> 4) * 8;
    v16h b0 = ldB(cur, bc, bk);
    v16h b1 = ldB(cur, bc + 16, bk);

    acc00 = __builtin_amdgcn_wmma_f32_16x16x32_f16(false, a0, false, b0, (short)0, acc00, false, false);
    acc01 = __builtin_amdgcn_wmma_f32_16x16x32_f16(false, a0, false, b1, (short)0, acc01, false, false);
    acc10 = __builtin_amdgcn_wmma_f32_16x16x32_f16(false, a1, false, b0, (short)0, acc10, false, false);
    acc11 = __builtin_amdgcn_wmma_f32_16x16x32_f16(false, a1, false, b1, (short)0, acc11, false, false);
  }

  // Epilogue: C/D layout (ISA): VGPR r -> M = r + 8*(lane>=16), N = lane%16.
  int nbase  = n0 + ntw * 32 + (lane & 15);
  int cobase = m0 + mt * 32 + (lane >> 4) * 8;
#pragma unroll
  for (int sn = 0; sn < 2; ++sn) {
    int gn = nbase + sn * 16;
    if (gn < N) {
      int p = p0 + gn;
      if (p < Ntot) {
        int b = p / OHW;
        int rem = p - b * OHW;
        float* o = outBase + (size_t)b * outBS + rem;
#pragma unroll
        for (int sm = 0; sm < 2; ++sm) {
          v8f av = (sm == 0) ? (sn == 0 ? acc00 : acc01)
                             : (sn == 0 ? acc10 : acc11);
          int cb = cobase + sm * 16;
#pragma unroll
          for (int r = 0; r < 8; ++r) {
            int co = cb + r;
            if (co < M) o[(size_t)co * OHW] = av[r];
          }
        }
      }
    }
  }
}

// ------------------------------ host drivers --------------------------------

struct Scratch {
  float* silu; float* rbf; float* mean; float* rstd;
  _Float16* w16; _Float16* col;
};

static void fastkan_conv(hipStream_t st, const float* in, int inBS, int Bn,
                         int Cin, int H, int W,
                         const float* bw, const float* sw, int cout,
                         int kd, int stride, int pad,
                         float* out, int outBS, bool applyAct, const Scratch& S) {
  int HW = H * W;
  int OH = (H + 2 * pad - kd) / stride + 1;
  int OW = (W + 2 * pad - kd) / stride + 1;
  const float* actB = in;
  int Cs = 0;
  if (applyAct) {
    int n = Bn * Cin * HW;
    k_silu<<<dim3((n + 255) / 256), 256, 0, st>>>(in, inBS, S.silu, Cin * HW, n);
    actB = S.silu;
    if (sw) {
      Cs = Cin * 8;
      k_instnorm_stats<<<dim3(Bn * Cin), 256, 0, st>>>(in, inBS, S.mean, S.rstd, Cin, HW);
      k_rbf<<<dim3((n + 255) / 256), 256, 0, st>>>(in, inBS, S.mean, S.rstd, S.rbf, Cin, HW, n);
    }
  }
  int ksz2 = kd * kd;
  int K = (Cin + Cs) * ksz2;
  int Kpad = (K + 31) & ~31;
  int Mpad = (cout + 63) & ~63;
  k_pack_w<<<dim3((Kpad + 255) / 256, Mpad), 256, 0, st>>>(
      bw, sw, S.w16, cout, Cin, Cs, ksz2, K, Kpad);

  int Ntot = Bn * OH * OW;
  int OHW = OH * OW;
  for (int p0 = 0; p0 < Ntot; p0 += PIXCHUNK) {
    int cn = Ntot - p0; if (cn > PIXCHUNK) cn = PIXCHUNK;
    k_im2col<<<dim3((Kpad + 255) / 256, cn), 256, 0, st>>>(
        actB, S.rbf, S.col, p0, Cin, Cs, H, W, OW, OHW, kd, stride, pad, K, Kpad);
    dim3 g((cn + BN - 1) / BN, Mpad / BM);
    k_gemm_wmma<<<g, 256, 0, st>>>(S.w16, S.col, out, cout, cn, Kpad, p0, OHW, outBS, Ntot);
  }
}

static const int BLK_CFG[4] = {6, 12, 24, 16};

extern "C" void kernel_launch(void* const* d_in, const int* in_sizes, int n_in,
                              void* d_out, int out_size, void* d_ws, size_t ws_size,
                              hipStream_t stream) {
  (void)n_in; (void)out_size; (void)ws_size;

  auto FP = [&](int i) { return (const float*)d_in[i]; };

  struct Kan { const float* b; const float* s; };
  struct Tr  { const float *gamma, *beta, *mean, *var, *w; };
  Kan c0{}; Kan L1[58]; Kan L2[58]; Tr tr[3];
  const float *n5gamma, *n5beta, *n5mean, *n5var, *clsW, *clsB, *x;

  int i = 0;
  bool xFirst = (in_sizes[0] == 2 * 3 * 224 * 224);   // dict-insertion order
  if (xFirst) {
    x = FP(i++);
    c0.b = FP(i++); c0.s = FP(i++);
    int l = 0;
    for (int bi = 0; bi < 4; ++bi)
      for (int li = 0; li < BLK_CFG[bi]; ++li, ++l) {
        L1[l].b = FP(i++); L1[l].s = FP(i++);
        L2[l].b = FP(i++); L2[l].s = FP(i++);
      }
    for (int t = 0; t < 3; ++t) {
      tr[t].gamma = FP(i++); tr[t].beta = FP(i++);
      tr[t].mean  = FP(i++); tr[t].var  = FP(i++); tr[t].w = FP(i++);
    }
    n5gamma = FP(i++); n5beta = FP(i++); n5mean = FP(i++); n5var = FP(i++);
    clsW = FP(i++); clsB = FP(i++);
  } else {                                             // jax sorted-key order
    int l = 0;
    for (int bi = 0; bi < 4; ++bi)
      for (int li = 0; li < BLK_CFG[bi]; ++li, ++l) {
        L1[l].b = FP(i++); L1[l].s = FP(i++);
        L2[l].b = FP(i++); L2[l].s = FP(i++);
      }
    clsB = FP(i++); clsW = FP(i++);
    c0.b = FP(i++); c0.s = FP(i++);
    n5beta = FP(i++); n5gamma = FP(i++); n5mean = FP(i++); n5var = FP(i++);
    for (int t = 0; t < 3; ++t) {
      tr[t].beta = FP(i++); tr[t].gamma = FP(i++);
      tr[t].mean = FP(i++); tr[t].var   = FP(i++); tr[t].w = FP(i++);
    }
    x = FP(i++);
  }

  // ---- workspace carve-up (bump allocator, 256B aligned) ----
  char* wp = (char*)d_ws;
  auto wsAlloc = [&](size_t bytes) -> void* {
    void* p = (void*)wp; wp += (bytes + 255) & ~(size_t)255; return p;
  };
  float*     concat = (float*)wsAlloc((size_t)1605632 * 4);   // max concat / conv0-out
  float*     tmp    = (float*)wsAlloc((size_t)1605632 * 4);   // bottleneck / trans-conv
  Scratch S;
  S.silu = (float*)wsAlloc((size_t)1605632 * 4);              // silu / bn buffer
  S.rbf  = (float*)wsAlloc((size_t)11239424 * 4);             // RBF (x8 channels)
  S.mean = (float*)wsAlloc((size_t)2048 * 4);
  S.rstd = (float*)wsAlloc((size_t)2048 * 4);
  S.w16  = (_Float16*)wsAlloc((size_t)1142784 * 2);           // packed f16 W (Mpad*Kpad max)
  S.col  = (_Float16*)wsAlloc((size_t)PIXCHUNK * 10368 * 2);  // im2col f16 chunk
  float* gapb = (float*)wsAlloc((size_t)2048 * 4);

  // ---- conv0 (FastKAN 7x7 s2 p3, 3->64) + maxpool 3x3 s2 p1 ----
  fastkan_conv(stream, x, 3 * 224 * 224, 2, 3, 224, 224,
               c0.b, c0.s, 64, 7, 2, 3, tmp, 64 * 112 * 112, true, S);
  {
    int n = 2 * 64 * 56 * 56;
    k_maxpool3<<<dim3((n + 255) / 256), 256, 0, stream>>>(
        tmp, concat, 256 * 56 * 56, 64, 112, 112, n);
  }

  // ---- dense blocks ----
  int sz = 56, nf = 64, l = 0;
  for (int bi = 0; bi < 4; ++bi) {
    int Cfull = nf + BLK_CFG[bi] * 32;
    int HWb = sz * sz;
    for (int li = 0; li < BLK_CFG[bi]; ++li, ++l) {
      int cin = nf + li * 32;
      // 1x1 FastKAN bottleneck: concat[:, :cin] -> tmp (128ch)
      fastkan_conv(stream, concat, Cfull * HWb, 2, cin, sz, sz,
                   L1[l].b, L1[l].s, 128, 1, 1, 0, tmp, 128 * HWb, true, S);
      // 3x3 FastKAN: tmp -> concat channel slice [nf+li*32, +32)
      fastkan_conv(stream, tmp, 128 * HWb, 2, 128, sz, sz,
                   L2[l].b, L2[l].s, 32, 3, 1, 1,
                   concat + (size_t)(nf + li * 32) * HWb, Cfull * HWb, true, S);
    }
    nf = Cfull;
    if (bi < 3) {
      // Transition: BN -> SELU -> 1x1 conv -> avgpool2
      int n = 2 * Cfull * HWb;
      k_bn_selu<<<dim3((n + 255) / 256), 256, 0, stream>>>(
          concat, tr[bi].gamma, tr[bi].beta, tr[bi].mean, tr[bi].var,
          S.silu, Cfull, HWb, n);
      fastkan_conv(stream, S.silu, Cfull * HWb, 2, Cfull, sz, sz,
                   tr[bi].w, nullptr, Cfull / 2, 1, 1, 0,
                   tmp, (Cfull / 2) * HWb, false, S);
      nf = Cfull / 2;
      int nsz = sz / 2;
      int nextCfull = nf + BLK_CFG[bi + 1] * 32;
      int n2 = 2 * nf * nsz * nsz;
      k_avgpool2<<<dim3((n2 + 255) / 256), 256, 0, stream>>>(
          tmp, concat, nextCfull * nsz * nsz, nf, sz, sz, n2);
      sz = nsz;
    }
  }

  // ---- norm5 + SELU + global avg pool + classifier ----
  {
    int HWb = sz * sz;                 // 7*7 = 49
    int n = 2 * nf * HWb;              // nf = 1024
    k_bn_selu<<<dim3((n + 255) / 256), 256, 0, stream>>>(
        concat, n5gamma, n5beta, n5mean, n5var, S.silu, nf, HWb, n);
    k_gap<<<dim3(2 * nf), 64, 0, stream>>>(S.silu, gapb, HWb);
    k_fc<<<dim3((2 * 1000 + 255) / 256), 256, 0, stream>>>(
        gapb, clsW, clsB, (float*)d_out, 2, 1000, nf);
  }
}
